// BaseSparseVAE_53961969107424
// MI455X (gfx1250) — matrix operand order repaired and checked by reading, not compiled, not run on hardware
//
#include <hip/hip_runtime.h>

typedef __attribute__((ext_vector_type(16))) __bf16 v16bf;
typedef __attribute__((ext_vector_type(2)))  __bf16 v2bf;
typedef __attribute__((ext_vector_type(8)))  float  v8f;

#define B_DIM 512
#define D_DIM 512
#define L_DIM 32
#define H_DIM 300
#define HP    320   // H padded to multiple of 32

// ---------- helpers ----------

union FragU { int i[8]; int4 q[2]; v16bf v; };
union PairU { __bf16 h[2]; v2bf v; unsigned int u; };

__device__ inline unsigned int cvt_pk_bf16_u32(float a, float b) {
  PairU p;
#if __has_builtin(__builtin_amdgcn_cvt_pk_bf16_f32)
  p.v = __builtin_amdgcn_cvt_pk_bf16_f32(a, b);
#else
  p.h[0] = (__bf16)a;
  p.h[1] = (__bf16)b;
#endif
  return p.u;
}

// Store a packed bf16 pair to two 16-bit locations: lo -> p0, hi -> p1 (d16_hi store).
__device__ inline void store_pk_split(__bf16* p0, __bf16* p1, float a, float b) {
  unsigned int w = cvt_pk_bf16_u32(a, b);
  *(short*)p0 = (short)(w & 0xffffu);
  *(short*)p1 = (short)(w >> 16);
}

// A fragment (16x32 bf16, M x K) from row-major bf16 [rows][ld]; ld, k0 even.
// ISA layout: lane = 16*hi + m; reg (4*seg+rr) holds K pair (k0 + 16*seg + 8*hi + 2*rr).
// Per lane the 8 dwords are two aligned 4-dword runs -> 2x b128.
__device__ inline v16bf load_a_frag(const __bf16* A, int row0, int k0, int ld, int lane) {
  const int* p = (const int*)A + (row0 + (lane & 15)) * (ld >> 1) + (k0 >> 1) + 4 * (lane >> 4);
  FragU u;
  u.q[0] = *(const int4*)p;
  u.q[1] = *(const int4*)(p + 8);
  return u.v;
}

// B fragment (32x16 bf16, K x N) from fragment-major packed weights:
// dst[((kt*NT + nt)*32 + lane)*8 + r] holds the bf16 pair (k,k+1) for this lane's n,
// with n = nt*16 + (lane&15), k = kt*32 + 8*(lane>>4) + 2*(r&3) + 16*(r>>2).  -> 2x b128 per lane.
template<int NPAD>
__device__ inline v16bf load_b_fragF(const int* Bp, int kt, int nt, int lane) {
  constexpr int NT = NPAD / 16;
  const int* p = Bp + (((kt * NT + nt) * 32 + lane) << 3);
  FragU u;
  u.q[0] = *(const int4*)p;
  u.q[1] = *(const int4*)(p + 4);
  return u.v;
}

__device__ inline v8f wmma_bf16(v16bf a, v16bf b, v8f c) {
  // (neg_a, A, neg_b, B, c_mod, C, reuse_a, reuse_b)
  return __builtin_amdgcn_wmma_f32_16x16x32_bf16(false, a, false, b, (short)0, c, false, false);
}

// ---------- prep kernels ----------

__global__ void convert_bf16_kernel(const float* __restrict__ src,
                                    __bf16* __restrict__ dst, int n) {
  int i = blockIdx.x * 256 + threadIdx.x;
  if (i < n) dst[i] = (__bf16)src[i];
}

// f32 row-major [K][N] -> bf16 fragment-major packed [Kpad/32][Npad/16][32 lanes][8 dwords].
__global__ void pack_weight_frag_kernel(const float* __restrict__ src, int* __restrict__ dst,
                                        int K, int N, int Kpad, int Npad) {
  int idx = blockIdx.x * 256 + threadIdx.x;
  int ntiles = Npad >> 4;
  int total = (Kpad >> 5) * ntiles * 256;
  if (idx >= total) return;
  int r  = idx & 7;
  int l  = (idx >> 3) & 31;
  int fi = idx >> 8;                 // kt*ntiles + nt
  int nt = fi % ntiles, kt = fi / ntiles;
  int n  = nt * 16 + (l & 15);
  int k  = kt * 32 + 8 * (l >> 4) + 2 * (r & 3) + 16 * (r >> 2);
  float f0 = (k < K && n < N) ? src[k * N + n] : 0.f;
  float f1 = (k + 1 < K && n < N) ? src[(k + 1) * N + n] : 0.f;
  dst[idx] = (int)cvt_pk_bf16_u32(f0, f1);
}

__global__ void pad_vec_kernel(const float* __restrict__ src, float* __restrict__ dst,
                               int N, int Npad) {
  int i = blockIdx.x * 256 + threadIdx.x;
  if (i < Npad) dst[i] = (i < N) ? src[i] : 0.f;
}

__global__ void pad_colw_kernel(const float* __restrict__ src, float* __restrict__ dst) {
  int i = blockIdx.x * 256 + threadIdx.x;   // D_DIM*HP
  if (i >= D_DIM * HP) return;
  int j = i / HP, n = i - j * HP;
  dst[i] = (n < H_DIM) ? src[j * H_DIM + n] : 0.f;
}

// ---------- generic WMMA GEMM: out = [relu](A[M,KPAD] @ B[KPAD,NPAD] + bias) ----------
// 128 threads = 4 waves; wave (wm,wn) computes a 32x32 C block (2x2 WMMA tiles).
template<int KPAD, int NPAD, bool RELU, bool TOBF16>
__global__ __launch_bounds__(128) void gemm_bf16_kernel(
    const __bf16* __restrict__ A, const int* __restrict__ Bp,
    const float* __restrict__ bias, __bf16* __restrict__ outb,
    float* __restrict__ outf, int M)
{
  int lane = threadIdx.x & 31;
  int wave = threadIdx.x >> 5;
  int row0 = blockIdx.y * 64 + (wave >> 1) * 32;
  int col0 = blockIdx.x * 64 + (wave & 1) * 32;
  if (col0 >= NPAD || row0 >= M) return;
  int nt0 = col0 >> 4;

  v8f cacc[2][2] = {};
#pragma unroll 1
  for (int kt = 0; kt < KPAD / 32; ++kt) {
    v16bf a0 = load_a_frag(A, row0,      kt * 32, KPAD, lane);
    v16bf a1 = load_a_frag(A, row0 + 16, kt * 32, KPAD, lane);
    v16bf b0 = load_b_fragF<NPAD>(Bp, kt, nt0,     lane);
    v16bf b1 = load_b_fragF<NPAD>(Bp, kt, nt0 + 1, lane);
    cacc[0][0] = wmma_bf16(a0, b0, cacc[0][0]);
    cacc[0][1] = wmma_bf16(a0, b1, cacc[0][1]);
    cacc[1][0] = wmma_bf16(a1, b0, cacc[1][0]);
    cacc[1][1] = wmma_bf16(a1, b1, cacc[1][1]);
  }

  int hi = lane >> 4, nloc = lane & 15;
#pragma unroll
  for (int ti = 0; ti < 2; ++ti) {
#pragma unroll
    for (int tj = 0; tj < 2; ++tj) {
      int n = col0 + tj * 16 + nloc;
      float bz = bias[n];
      int rowb = row0 + ti * 16 + 8 * hi;
#pragma unroll
      for (int r = 0; r < 8; r += 2) {
        float v0 = cacc[ti][tj][r] + bz;
        float v1 = cacc[ti][tj][r + 1] + bz;
        if (RELU) { v0 = fmaxf(v0, 0.f); v1 = fmaxf(v1, 0.f); }
        if (TOBF16) {
          store_pk_split(&outb[(rowb + r) * NPAD + n],
                         &outb[(rowb + r + 1) * NPAD + n], v0, v1);
        } else {
          outf[(rowb + r) * NPAD + n]     = v0;
          outf[(rowb + r + 1) * NPAD + n] = v1;
        }
      }
    }
  }
}

// ---------- z = z_mean + eps * exp(0.5*z_log_var) (reads/writes d_out slots) ----------
__global__ void z_kernel(const float* __restrict__ eps, float* __restrict__ out) {
  int i = blockIdx.x * 256 + threadIdx.x;
  if (i >= B_DIM * L_DIM) return;
  const int ZOFF = B_DIM * D_DIM;              // 262144
  float zm = out[ZOFF + 16384 + i];            // z_mean slot
  float zl = out[ZOFF + 2 * 16384 + i];        // z_log_var slot
  out[ZOFF + i] = zm + eps[i] * expf(0.5f * zl);
}

// ---------- fused per-column decoder ----------
// grid = (4 B-tiles of 128, 512 columns j); 128 threads = 4 waves.
// Each wave owns TWO 16-row M tiles so every B fragment feeds 2 WMMAs.
__global__ __launch_bounds__(128) void decoder_kernel(
    const float* __restrict__ zf,       // [B,32] f32 (d_out z slot)
    const float* __restrict__ Wm,       // [D,32] mask
    const int* __restrict__ g1p,        // gen_w1 frag-packed [1][20][32][8]
    const int* __restrict__ g2p,        // gen_w2 frag-packed [10][20][32][8]
    const float* __restrict__ gb2p,     // [HP] (zero padded)
    const float* __restrict__ colwp,    // [D][HP] (zero padded)
    const float* __restrict__ colb,     // [D]
    float* __restrict__ xmean)          // [B,D]
{
  __shared__ __bf16 sA[128 * 32];     // masked z tile (bf16)   8 KB
  __shared__ __bf16 sG1[128 * HP];    // g1 tile (bf16)        80 KB

  const int j    = blockIdx.y;
  const int b0   = blockIdx.x * 128;
  const int tid  = threadIdx.x;
  const int lane = tid & 31;
  const int wave = tid >> 5;
  const int wrow = wave * 32;                 // two tiles: wrow, wrow+16
  const int hi   = lane >> 4, nloc = lane & 15;

  // phase 0: masked[b,l] = z[b,l] * W[j,l]
  for (int e = tid; e < 128 * 32; e += 128) {
    int b = e >> 5, l = e & 31;
    sA[e] = (__bf16)(zf[(b0 + b) * 32 + l] * Wm[j * 32 + l]);
  }
  __syncthreads();

  // phase 1: g1 = relu(masked @ gen_w1)   (K = 32, single k-step; no bias)
  {
    v16bf a0 = load_a_frag(sA, wrow,      0, 32, lane);
    v16bf a1 = load_a_frag(sA, wrow + 16, 0, 32, lane);
#pragma unroll 1
    for (int nt = 0; nt < HP / 16; ++nt) {
      v16bf b = load_b_fragF<HP>(g1p, 0, nt, lane);
      v8f c0 = {}, c1 = {};
      c0 = wmma_bf16(a0, b, c0);
      c1 = wmma_bf16(a1, b, c1);
      int n = nt * 16 + nloc;
      __bf16* q0 = &sG1[(wrow + 8 * hi) * HP + n];
      __bf16* q1 = &sG1[(wrow + 16 + 8 * hi) * HP + n];
#pragma unroll
      for (int r = 0; r < 8; r += 2) {
        store_pk_split(q0 + r * HP, q0 + (r + 1) * HP,
                       fmaxf(c0[r], 0.f), fmaxf(c0[r + 1], 0.f));
        store_pk_split(q1 + r * HP, q1 + (r + 1) * HP,
                       fmaxf(c1[r], 0.f), fmaxf(c1[r + 1], 0.f));
      }
    }
  }
  __syncthreads();

  // phase 2: x_rec[j,b] = sum_h relu(g1 @ gen_w2 + b2)[b,h] * col_w[j,h] + col_b[j]
  v16bf aq0[HP / 32], aq1[HP / 32];
#pragma unroll
  for (int ks = 0; ks < HP / 32; ++ks) {
    aq0[ks] = load_a_frag(sG1, wrow,      ks * 32, HP, lane);
    aq1[ks] = load_a_frag(sG1, wrow + 16, ks * 32, HP, lane);
  }

  float acc0[8], acc1[8];
#pragma unroll
  for (int r = 0; r < 8; ++r) { acc0[r] = 0.f; acc1[r] = 0.f; }

#pragma unroll 1
  for (int nt = 0; nt < HP / 16; ++nt) {
    v8f c0 = {}, c1 = {};
#pragma unroll
    for (int ks = 0; ks < HP / 32; ++ks) {
      v16bf b = load_b_fragF<HP>(g2p, ks, nt, lane);
      c0 = wmma_bf16(aq0[ks], b, c0);
      c1 = wmma_bf16(aq1[ks], b, c1);
    }
    int n = nt * 16 + nloc;
    float bz = gb2p[n];
    float cw = colwp[j * HP + n];
#pragma unroll
    for (int r = 0; r < 8; ++r) {
      acc0[r] += fmaxf(c0[r] + bz, 0.f) * cw;
      acc1[r] += fmaxf(c1[r] + bz, 0.f) * cw;
    }
  }

  // reduce over the 16 lanes of each half (each lane holds one N column per tile)
#pragma unroll
  for (int r = 0; r < 8; ++r) {
    float s0 = acc0[r], s1 = acc1[r];
    s0 += __shfl_xor(s0, 1, 16); s1 += __shfl_xor(s1, 1, 16);
    s0 += __shfl_xor(s0, 2, 16); s1 += __shfl_xor(s1, 2, 16);
    s0 += __shfl_xor(s0, 4, 16); s1 += __shfl_xor(s1, 4, 16);
    s0 += __shfl_xor(s0, 8, 16); s1 += __shfl_xor(s1, 8, 16);
    acc0[r] = s0; acc1[r] = s1;
  }
  if (nloc == 0) {
    float cb = colb[j];
#pragma unroll
    for (int r = 0; r < 8; ++r) {
      int row = b0 + wrow + r + 8 * hi;
      xmean[row * D_DIM + j]        = acc0[r] + cb;
      xmean[(row + 16) * D_DIM + j] = acc1[r] + cb;
    }
  }
}

// ---------- launcher ----------

extern "C" void kernel_launch(void* const* d_in, const int* in_sizes, int n_in,
                              void* d_out, int out_size, void* d_ws, size_t ws_size,
                              hipStream_t stream) {
  const float* x      = (const float*)d_in[0];
  const float* eps    = (const float*)d_in[1];
  const float* Wm     = (const float*)d_in[2];
  const float* qz_w1  = (const float*)d_in[3];
  const float* qz_b1  = (const float*)d_in[4];
  const float* qz_w2  = (const float*)d_in[5];
  const float* qz_b2  = (const float*)d_in[6];
  const float* zm_w   = (const float*)d_in[7];
  const float* zm_b   = (const float*)d_in[8];
  const float* zl_w   = (const float*)d_in[9];
  const float* zl_b   = (const float*)d_in[10];
  const float* gen_w1 = (const float*)d_in[11];
  const float* gen_w2 = (const float*)d_in[12];
  const float* gen_b2 = (const float*)d_in[13];
  const float* col_w  = (const float*)d_in[14];
  const float* col_b  = (const float*)d_in[15];
  float* out = (float*)d_out;

  char* ws = (char*)d_ws;
  size_t off = 0;
  auto alloc = [&](size_t bytes) -> void* {
    void* p = ws + off;
    off = (off + bytes + 255) & ~(size_t)255;
    return p;
  };

  __bf16* xb   = (__bf16*)alloc(512 * 512 * 2);
  int*   w1p   = (int*)alloc((size_t)256 * HP * 4);   // qz_w1: [16][20][32][8]
  int*   w2p   = (int*)alloc((size_t)160 * HP * 4);   // qz_w2: [10][20][32][8]
  int*   zmp   = (int*)alloc((size_t)160 * 32 * 4);   // zm_w : [10][2][32][8]
  int*   zlp   = (int*)alloc((size_t)160 * 32 * 4);
  int*   g1p   = (int*)alloc((size_t)16  * HP * 4);   // gen_w1: [1][20][32][8]
  int*   g2p   = (int*)alloc((size_t)160 * HP * 4);   // gen_w2: [10][20][32][8]
  float* b1p   = (float*)alloc(HP * 4);
  float* b2p   = (float*)alloc(HP * 4);
  float* gb2p  = (float*)alloc(HP * 4);
  float* colwp = (float*)alloc((size_t)D_DIM * HP * 4);
  __bf16* h1   = (__bf16*)alloc((size_t)512 * HP * 2);
  __bf16* h2   = (__bf16*)alloc((size_t)512 * HP * 2);

  // prep
  convert_bf16_kernel<<<(512 * 512 + 255) / 256, 256, 0, stream>>>(x, xb, 512 * 512);
  pack_weight_frag_kernel<<<(256 * HP + 255) / 256, 256, 0, stream>>>(qz_w1, w1p, 512, 300, 512, HP);
  pack_weight_frag_kernel<<<(160 * HP + 255) / 256, 256, 0, stream>>>(qz_w2, w2p, 300, 300, HP, HP);
  pack_weight_frag_kernel<<<(160 * 32 + 255) / 256, 256, 0, stream>>>(zm_w, zmp, 300, 32, HP, 32);
  pack_weight_frag_kernel<<<(160 * 32 + 255) / 256, 256, 0, stream>>>(zl_w, zlp, 300, 32, HP, 32);
  pack_weight_frag_kernel<<<(16 * HP + 255) / 256, 256, 0, stream>>>(gen_w1, g1p, 32, 300, 32, HP);
  pack_weight_frag_kernel<<<(160 * HP + 255) / 256, 256, 0, stream>>>(gen_w2, g2p, 300, 300, HP, HP);
  pad_vec_kernel<<<2, 256, 0, stream>>>(qz_b1, b1p, 300, HP);
  pad_vec_kernel<<<2, 256, 0, stream>>>(qz_b2, b2p, 300, HP);
  pad_vec_kernel<<<2, 256, 0, stream>>>(gen_b2, gb2p, 300, HP);
  pad_colw_kernel<<<(D_DIM * HP + 255) / 256, 256, 0, stream>>>(col_w, colwp);

  // encoder
  gemm_bf16_kernel<512, HP, true, true><<<dim3(HP / 64, 512 / 64), 128, 0, stream>>>(
      xb, w1p, b1p, h1, nullptr, 512);
  gemm_bf16_kernel<HP, HP, true, true><<<dim3(HP / 64, 512 / 64), 128, 0, stream>>>(
      h1, w2p, b2p, h2, nullptr, 512);
  gemm_bf16_kernel<HP, 32, false, false><<<dim3(1, 512 / 64), 128, 0, stream>>>(
      h2, zmp, zm_b, nullptr, out + 262144 + 16384, 512);       // z_mean
  gemm_bf16_kernel<HP, 32, false, false><<<dim3(1, 512 / 64), 128, 0, stream>>>(
      h2, zlp, zl_b, nullptr, out + 262144 + 2 * 16384, 512);   // z_log_var
  z_kernel<<<(512 * 32 + 255) / 256, 256, 0, stream>>>(eps, out);

  // fused decoder
  decoder_kernel<<<dim3(512 / 128, D_DIM), 128, 0, stream>>>(
      out + 262144, Wm, g1p, g2p, gb2p, colwp, col_b, out);
}
